// DecoderCell_81432579932381
// MI455X (gfx1250) — matrix operand order repaired and checked by reading, not compiled, not run on hardware
//
#include <hip/hip_runtime.h>

typedef __bf16 bf16_t;
typedef bf16_t v16bf __attribute__((ext_vector_type(16)));
typedef bf16_t v8bf  __attribute__((ext_vector_type(8)));
typedef float  v8f   __attribute__((ext_vector_type(8)));

#define N_GEN 512
#define N_CON 256
#define N_CO  128
#define N_FAC 128
#define N_CI  128
#define ROW   1280
#define O_GEN 0
#define O_CON 512
#define O_CM  768
#define O_CL  896
#define O_GI  1024
#define O_F   1152

__device__ __forceinline__ float fsigmoid(float x) { return 1.0f / (1.0f + __expf(-x)); }
__device__ __forceinline__ float ftanh(float x) {
  float e = __expf(2.0f * x);
  return (e - 1.0f) / (e + 1.0f);
}
__device__ __forceinline__ float fclip(float x) { return fminf(fmaxf(x, -5.0f), 5.0f); }
__device__ __forceinline__ v8f vzero() {
  v8f z;
#pragma unroll
  for (int i = 0; i < 8; ++i) z[i] = 0.0f;
  return z;
}

// A fragment: 16x32 bf16 tile from LDS (row-major, stride ld halves).
// lane<16: M=lane, K pairs kb+{0..7} and kb+{16..23}; lane>=16: kb += 8.
__device__ __forceinline__ v16bf lds_A(const bf16_t* base, int ld, int mbase, int k, int lane) {
  const bf16_t* p = base + (mbase + (lane & 15)) * ld + k + ((lane & 16) ? 8 : 0);
  v8bf lo = *(const v8bf*)p;
  v8bf hi = *(const v8bf*)(p + 16);
  v16bf a;
#pragma unroll
  for (int i = 0; i < 8; ++i) { a[i] = lo[i]; a[i + 8] = hi[i]; }
  return a;
}

// B fragment: 32x16 bf16, B[k][n] = W[n][k] with W row-major [n][ldw].
// lane<16: N=lane, K = k+{0..7} and k+{16..23}; lane>=16: k += 8.
__device__ __forceinline__ v16bf glb_B(const bf16_t* w, int ldw, int nbase, int k, int lane) {
  const bf16_t* p = w + (size_t)(nbase + (lane & 15)) * ldw + k + ((lane & 16) ? 8 : 0);
  v8bf lo = *(const v8bf*)p;
  v8bf hi = *(const v8bf*)(p + 16);
  v16bf b;
#pragma unroll
  for (int i = 0; i < 8; ++i) { b[i] = lo[i]; b[i + 8] = hi[i]; }
  return b;
}

__device__ __forceinline__ v8f wmma_bf(v16bf a, v16bf b, v8f c) {
  return __builtin_amdgcn_wmma_f32_16x16x32_bf16(false, a, false, b, (short)0, c, false, false);
}

// ---------------- prep kernels ----------------
__global__ void k_cvt(const float* __restrict__ s, bf16_t* __restrict__ d, int n) {
  int i = blockIdx.x * blockDim.x + threadIdx.x;
  if (i < n) d[i] = (bf16_t)s[i];
}

// fac_w (128 x 512): normalize each column (axis 0) by its L2 norm, emit bf16.
__global__ void k_facnorm(const float* __restrict__ fw, bf16_t* __restrict__ dst) {
  __shared__ float nrm[N_GEN];
  int t = threadIdx.x;  // 512 threads
  float s = 0.0f;
  for (int n = 0; n < N_FAC; ++n) { float v = fw[n * N_GEN + t]; s += v * v; }
  nrm[t] = fmaxf(sqrtf(s), 1e-12f);
  __syncthreads();
  for (int i = t; i < N_FAC * N_GEN; i += N_GEN) {
    int k = i & (N_GEN - 1);
    dst[i] = (bf16_t)(fw[i] / nrm[k]);
  }
}

// ---------------- controller GRU + co projection ----------------
// 32 rows/block, 8 waves; wave w owns 32 gate columns [32w, 32w+32) of CON.
__global__ __launch_bounds__(256) void k_con_co(
    const float* __restrict__ x, const float* __restrict__ h0,
    const bf16_t* __restrict__ wih, const float* __restrict__ bih,
    const bf16_t* __restrict__ whh, const float* __restrict__ bhh,
    const bf16_t* __restrict__ cow, const float* __restrict__ cob,
    float* __restrict__ out) {
  constexpr int LD = 264;  // 264 halves = 528B: 132 dwords == 4 mod 64 banks
  __shared__ __attribute__((aligned(16))) bf16_t sX[32 * LD];   // con_input
  __shared__ __attribute__((aligned(16))) bf16_t sH[32 * LD];   // h, then h'
  __shared__ __attribute__((aligned(16))) bf16_t sRH[32 * LD];  // r*h
  const int tid = threadIdx.x;
  const int lane = tid & 31, wid = tid >> 5;
  const int row0 = blockIdx.x * 32;

  for (int i = tid; i < 32 * 128; i += 256) {
    int r = i >> 7, c = i & 127;
    sX[r * LD + c]       = (bf16_t)x[(size_t)(row0 + r) * N_CI + c];
    sX[r * LD + 128 + c] = (bf16_t)h0[(size_t)(row0 + r) * ROW + O_F + c];
  }
  for (int i = tid; i < 32 * 256; i += 256) {
    int r = i >> 8, c = i & 255;
    sH[r * LD + c] = (bf16_t)h0[(size_t)(row0 + r) * ROW + O_CON + c];
  }
  __syncthreads();

  const int jc = wid * 32;

  // ---- phase 1: z, r (input GEMM + hidden GEMM share accumulators) ----
  v8f az[2][2], ar[2][2];
#pragma unroll
  for (int mt = 0; mt < 2; ++mt)
#pragma unroll
    for (int nt = 0; nt < 2; ++nt) { az[mt][nt] = vzero(); ar[mt][nt] = vzero(); }

  for (int k = 0; k < 256; k += 32) {
    v16bf a0 = lds_A(sX, LD, 0, k, lane);
    v16bf a1 = lds_A(sX, LD, 16, k, lane);
#pragma unroll
    for (int nt = 0; nt < 2; ++nt) {
      v16bf bz = glb_B(wih, 256, jc + nt * 16, k, lane);
      v16bf br = glb_B(wih, 256, 256 + jc + nt * 16, k, lane);
      az[0][nt] = wmma_bf(a0, bz, az[0][nt]);
      az[1][nt] = wmma_bf(a1, bz, az[1][nt]);
      ar[0][nt] = wmma_bf(a0, br, ar[0][nt]);
      ar[1][nt] = wmma_bf(a1, br, ar[1][nt]);
    }
  }
  for (int k = 0; k < 256; k += 32) {
    v16bf a0 = lds_A(sH, LD, 0, k, lane);
    v16bf a1 = lds_A(sH, LD, 16, k, lane);
#pragma unroll
    for (int nt = 0; nt < 2; ++nt) {
      v16bf bz = glb_B(whh, 256, jc + nt * 16, k, lane);
      v16bf br = glb_B(whh, 256, 256 + jc + nt * 16, k, lane);
      az[0][nt] = wmma_bf(a0, bz, az[0][nt]);
      az[1][nt] = wmma_bf(a1, bz, az[1][nt]);
      ar[0][nt] = wmma_bf(a0, br, ar[0][nt]);
      ar[1][nt] = wmma_bf(a1, br, ar[1][nt]);
    }
  }

  float zreg[2][2][8];
#pragma unroll
  for (int nt = 0; nt < 2; ++nt) {
    int j = jc + nt * 16 + (lane & 15);
    float bz = bih[j] + bhh[j];
    float br = bih[256 + j] + bhh[256 + j];
#pragma unroll
    for (int mt = 0; mt < 2; ++mt)
#pragma unroll
      for (int v = 0; v < 8; ++v) {
        int m = mt * 16 + v + ((lane & 16) ? 8 : 0);
        float z = fsigmoid(az[mt][nt][v] + bz);
        float r = fsigmoid(ar[mt][nt][v] + br);
        zreg[mt][nt][v] = z;
        float h = (float)sH[m * LD + j];
        sRH[m * LD + j] = (bf16_t)(r * h);  // candidate uses (r*h) @ W_hn^T
      }
  }
  __syncthreads();

  // ---- phase 2: n gate, new h ----
  v8f an[2][2];
#pragma unroll
  for (int mt = 0; mt < 2; ++mt)
#pragma unroll
    for (int nt = 0; nt < 2; ++nt) an[mt][nt] = vzero();

  for (int k = 0; k < 256; k += 32) {
    v16bf a0 = lds_A(sX, LD, 0, k, lane);
    v16bf a1 = lds_A(sX, LD, 16, k, lane);
#pragma unroll
    for (int nt = 0; nt < 2; ++nt) {
      v16bf bn = glb_B(wih, 256, 512 + jc + nt * 16, k, lane);
      an[0][nt] = wmma_bf(a0, bn, an[0][nt]);
      an[1][nt] = wmma_bf(a1, bn, an[1][nt]);
    }
  }
  for (int k = 0; k < 256; k += 32) {
    v16bf a0 = lds_A(sRH, LD, 0, k, lane);
    v16bf a1 = lds_A(sRH, LD, 16, k, lane);
#pragma unroll
    for (int nt = 0; nt < 2; ++nt) {
      v16bf bn = glb_B(whh, 256, 512 + jc + nt * 16, k, lane);
      an[0][nt] = wmma_bf(a0, bn, an[0][nt]);
      an[1][nt] = wmma_bf(a1, bn, an[1][nt]);
    }
  }
#pragma unroll
  for (int nt = 0; nt < 2; ++nt) {
    int j = jc + nt * 16 + (lane & 15);
    float bn = bih[512 + j] + bhh[512 + j];
#pragma unroll
    for (int mt = 0; mt < 2; ++mt)
#pragma unroll
      for (int v = 0; v < 8; ++v) {
        int m = mt * 16 + v + ((lane & 16) ? 8 : 0);
        float n = ftanh(an[mt][nt][v] + bn);
        float z = zreg[mt][nt][v];
        float h = (float)sH[m * LD + j];
        float hn = fclip(z * h + (1.0f - z) * n);
        out[(size_t)(row0 + m) * ROW + O_CON + j] = hn;
        sH[m * LD + j] = (bf16_t)hn;  // in-place: phase 3 A-matrix
      }
  }
  __syncthreads();

  // ---- phase 3: co projection (mean | logstd) ----
  v8f ac[2][2];
#pragma unroll
  for (int mt = 0; mt < 2; ++mt)
#pragma unroll
    for (int nt = 0; nt < 2; ++nt) ac[mt][nt] = vzero();
  for (int k = 0; k < 256; k += 32) {
    v16bf a0 = lds_A(sH, LD, 0, k, lane);
    v16bf a1 = lds_A(sH, LD, 16, k, lane);
#pragma unroll
    for (int nt = 0; nt < 2; ++nt) {
      v16bf b = glb_B(cow, 256, jc + nt * 16, k, lane);
      ac[0][nt] = wmma_bf(a0, b, ac[0][nt]);
      ac[1][nt] = wmma_bf(a1, b, ac[1][nt]);
    }
  }
#pragma unroll
  for (int nt = 0; nt < 2; ++nt) {
    int n = jc + nt * 16 + (lane & 15);
    float bb = cob[n];
#pragma unroll
    for (int mt = 0; mt < 2; ++mt)
#pragma unroll
      for (int v = 0; v < 8; ++v) {
        int m = mt * 16 + v + ((lane & 16) ? 8 : 0);
        float val = ac[mt][nt][v] + bb;
        size_t ro = (size_t)(row0 + m) * ROW;
        if (n < 128) { out[ro + O_CM + n] = val; out[ro + O_GI + n] = val; }
        else          { out[ro + O_CL + (n - 128)] = val; }
      }
  }
}

// ---------------- generator GRU + factor projection ----------------
// 32 rows/block, 8 waves; wave w owns 64 gen columns [64w, 64w+64).
__global__ __launch_bounds__(256) void k_gen_fac(
    const float* __restrict__ h0,
    const bf16_t* __restrict__ wih, const float* __restrict__ bih,
    const bf16_t* __restrict__ whh, const float* __restrict__ bhh,
    const bf16_t* __restrict__ fw, float* __restrict__ out) {
  constexpr int LDH = 520, LDI = 136;
  extern __shared__ bf16_t smem[];
  bf16_t* sXg = smem;             // 32*LDI : gen_input (co_mean)
  bf16_t* sHg = smem + 32 * LDI;  // 32*LDH : h, then h'
  bf16_t* sRH = sHg + 32 * LDH;   // 32*LDH : r*h
  const int tid = threadIdx.x, lane = tid & 31, wid = tid >> 5;
  const int row0 = blockIdx.x * 32;

  for (int i = tid; i < 32 * 128; i += 256) {
    int r = i >> 7, c = i & 127;
    sXg[r * LDI + c] = (bf16_t)out[(size_t)(row0 + r) * ROW + O_CM + c];
  }
  for (int i = tid; i < 32 * 512; i += 256) {
    int r = i >> 9, c = i & 511;
    sHg[r * LDH + c] = (bf16_t)h0[(size_t)(row0 + r) * ROW + O_GEN + c];
  }
  __syncthreads();

  const int jc = wid * 64;

  // ---- phase 1: z, r ----
  v8f az[2][4], ar[2][4];
#pragma unroll
  for (int mt = 0; mt < 2; ++mt)
#pragma unroll
    for (int nt = 0; nt < 4; ++nt) { az[mt][nt] = vzero(); ar[mt][nt] = vzero(); }

  for (int k = 0; k < 128; k += 32) {
    v16bf a0 = lds_A(sXg, LDI, 0, k, lane);
    v16bf a1 = lds_A(sXg, LDI, 16, k, lane);
#pragma unroll
    for (int nt = 0; nt < 4; ++nt) {
      v16bf bz = glb_B(wih, 128, jc + nt * 16, k, lane);
      v16bf br = glb_B(wih, 128, 512 + jc + nt * 16, k, lane);
      az[0][nt] = wmma_bf(a0, bz, az[0][nt]);
      az[1][nt] = wmma_bf(a1, bz, az[1][nt]);
      ar[0][nt] = wmma_bf(a0, br, ar[0][nt]);
      ar[1][nt] = wmma_bf(a1, br, ar[1][nt]);
    }
  }
  for (int k = 0; k < 512; k += 32) {
    v16bf a0 = lds_A(sHg, LDH, 0, k, lane);
    v16bf a1 = lds_A(sHg, LDH, 16, k, lane);
#pragma unroll
    for (int nt = 0; nt < 4; ++nt) {
      v16bf bz = glb_B(whh, 512, jc + nt * 16, k, lane);
      v16bf br = glb_B(whh, 512, 512 + jc + nt * 16, k, lane);
      az[0][nt] = wmma_bf(a0, bz, az[0][nt]);
      az[1][nt] = wmma_bf(a1, bz, az[1][nt]);
      ar[0][nt] = wmma_bf(a0, br, ar[0][nt]);
      ar[1][nt] = wmma_bf(a1, br, ar[1][nt]);
    }
  }

  float zreg[2][4][8];
#pragma unroll
  for (int nt = 0; nt < 4; ++nt) {
    int j = jc + nt * 16 + (lane & 15);
    float bz = bih[j] + bhh[j];
    float br = bih[512 + j] + bhh[512 + j];
#pragma unroll
    for (int mt = 0; mt < 2; ++mt)
#pragma unroll
      for (int v = 0; v < 8; ++v) {
        int m = mt * 16 + v + ((lane & 16) ? 8 : 0);
        float z = fsigmoid(az[mt][nt][v] + bz);
        float r = fsigmoid(ar[mt][nt][v] + br);
        zreg[mt][nt][v] = z;
        float h = (float)sHg[m * LDH + j];
        sRH[m * LDH + j] = (bf16_t)(r * h);
      }
  }
  __syncthreads();

  // ---- phase 2: n gate, new gen_state ----
  v8f an[2][4];
#pragma unroll
  for (int mt = 0; mt < 2; ++mt)
#pragma unroll
    for (int nt = 0; nt < 4; ++nt) an[mt][nt] = vzero();

  for (int k = 0; k < 128; k += 32) {
    v16bf a0 = lds_A(sXg, LDI, 0, k, lane);
    v16bf a1 = lds_A(sXg, LDI, 16, k, lane);
#pragma unroll
    for (int nt = 0; nt < 4; ++nt) {
      v16bf bn = glb_B(wih, 128, 1024 + jc + nt * 16, k, lane);
      an[0][nt] = wmma_bf(a0, bn, an[0][nt]);
      an[1][nt] = wmma_bf(a1, bn, an[1][nt]);
    }
  }
  for (int k = 0; k < 512; k += 32) {
    v16bf a0 = lds_A(sRH, LDH, 0, k, lane);
    v16bf a1 = lds_A(sRH, LDH, 16, k, lane);
#pragma unroll
    for (int nt = 0; nt < 4; ++nt) {
      v16bf bn = glb_B(whh, 512, 1024 + jc + nt * 16, k, lane);
      an[0][nt] = wmma_bf(a0, bn, an[0][nt]);
      an[1][nt] = wmma_bf(a1, bn, an[1][nt]);
    }
  }
#pragma unroll
  for (int nt = 0; nt < 4; ++nt) {
    int j = jc + nt * 16 + (lane & 15);
    float bn = bih[1024 + j] + bhh[1024 + j];
#pragma unroll
    for (int mt = 0; mt < 2; ++mt)
#pragma unroll
      for (int v = 0; v < 8; ++v) {
        int m = mt * 16 + v + ((lane & 16) ? 8 : 0);
        float n = ftanh(an[mt][nt][v] + bn);
        float z = zreg[mt][nt][v];
        float h = (float)sHg[m * LDH + j];
        float hn = fclip(z * h + (1.0f - z) * n);
        out[(size_t)(row0 + m) * ROW + O_GEN + j] = hn;
        sHg[m * LDH + j] = (bf16_t)hn;  // in-place: factor A-matrix
      }
  }
  __syncthreads();

  // ---- phase 3: factor = gen_state @ w_norm^T (wave owns 16 cols) ----
  const int nc = wid * 16;
  v8f af[2];
  af[0] = vzero(); af[1] = vzero();
  for (int k = 0; k < 512; k += 32) {
    v16bf a0 = lds_A(sHg, LDH, 0, k, lane);
    v16bf a1 = lds_A(sHg, LDH, 16, k, lane);
    v16bf b = glb_B(fw, 512, nc, k, lane);
    af[0] = wmma_bf(a0, b, af[0]);
    af[1] = wmma_bf(a1, b, af[1]);
  }
  {
    int n = nc + (lane & 15);
#pragma unroll
    for (int mt = 0; mt < 2; ++mt)
#pragma unroll
      for (int v = 0; v < 8; ++v) {
        int m = mt * 16 + v + ((lane & 16) ? 8 : 0);
        out[(size_t)(row0 + m) * ROW + O_F + n] = af[mt][v];
      }
  }
}

extern "C" void kernel_launch(void* const* d_in, const int* in_sizes, int n_in,
                              void* d_out, int out_size, void* d_ws, size_t ws_size,
                              hipStream_t stream) {
  const float* x        = (const float*)d_in[0];
  const float* h0       = (const float*)d_in[1];
  const float* con_w_ih = (const float*)d_in[2];
  const float* con_b_ih = (const float*)d_in[3];
  const float* con_w_hh = (const float*)d_in[4];
  const float* con_b_hh = (const float*)d_in[5];
  const float* co_w     = (const float*)d_in[6];
  const float* co_b     = (const float*)d_in[7];
  const float* gen_w_ih = (const float*)d_in[8];
  const float* gen_b_ih = (const float*)d_in[9];
  const float* gen_w_hh = (const float*)d_in[10];
  const float* gen_b_hh = (const float*)d_in[11];
  const float* fac_w    = (const float*)d_in[12];
  float* out = (float*)d_out;

  // bf16 weight workspace layout (elements)
  bf16_t* ws = (bf16_t*)d_ws;
  bf16_t* w_conih = ws;                 // 768*256
  bf16_t* w_conhh = w_conih + 196608;   // 768*256
  bf16_t* w_co    = w_conhh + 196608;   // 256*256
  bf16_t* w_genih = w_co + 65536;       // 1536*128
  bf16_t* w_genhh = w_genih + 196608;   // 1536*512
  bf16_t* w_fac   = w_genhh + 786432;   // 128*512 (normalized)

  k_cvt<<<(768 * 256 + 255) / 256, 256, 0, stream>>>(con_w_ih, w_conih, 768 * 256);
  k_cvt<<<(768 * 256 + 255) / 256, 256, 0, stream>>>(con_w_hh, w_conhh, 768 * 256);
  k_cvt<<<(256 * 256 + 255) / 256, 256, 0, stream>>>(co_w, w_co, 256 * 256);
  k_cvt<<<(1536 * 128 + 255) / 256, 256, 0, stream>>>(gen_w_ih, w_genih, 1536 * 128);
  k_cvt<<<(1536 * 512 + 255) / 256, 256, 0, stream>>>(gen_w_hh, w_genhh, 1536 * 512);
  k_facnorm<<<1, 512, 0, stream>>>(fac_w, w_fac);

  const int B = in_sizes[0] / N_CI;  // 16384
  const int nblk = B / 32;

  k_con_co<<<nblk, 256, 0, stream>>>(x, h0, w_conih, con_b_ih, w_conhh, con_b_hh,
                                     w_co, co_b, out);

  const size_t smem3 = (size_t)(32 * 136 + 2 * 32 * 520) * sizeof(bf16_t);  // 75264 B
  k_gen_fac<<<nblk, 256, smem3, stream>>>(h0, w_genih, gen_b_ih, w_genhh, gen_b_hh,
                                          w_fac, out);
}
